// CPCModel_6133213298846
// MI455X (gfx1250) — matrix operand order repaired and checked by reading, not compiled, not run on hardware
//
#include <hip/hip_runtime.h>
#include <hip/hip_bf16.h>

typedef _Float16 half_t;
typedef __attribute__((ext_vector_type(16))) _Float16 v16h;
typedef __attribute__((ext_vector_type(8)))  _Float16 v8h;
typedef __attribute__((ext_vector_type(8)))  float    v8f;

#define MT   64
#define NT   32
#define KT2  64          // K-step for the generic / conv3x3 kernels (2 WMMAs)
#define LDA  72          // padded LDS row stride (halves) for K=64 tiles
#define KTE  32          // K-step for the encoder kernel (K=96)
#define LDE  40          // padded LDS row stride (halves) for K=32 tiles

// ---------------- CDNA5 wave32 WMMA fragment loaders (LDS -> VGPR) -----------
// A: 16x32 f16. lanes 0-15: M=lane, halves[0..7]=K0..7, halves[8..15]=K16..23
//               lanes 16-31: M=lane-16, halves[0..7]=K8..15, [8..15]=K24..31
template <int LDK>
__device__ __forceinline__ v16h frag_a(const half_t* base) {
  const int lane = threadIdx.x & 31;
  const half_t* p = base + (lane & 15) * LDK + ((lane >> 4) << 3);
  v8h lo = *(const v8h*)(p);
  v8h hi = *(const v8h*)(p + 16);
  v16h f;
#pragma unroll
  for (int i = 0; i < 8; ++i) { f[i] = lo[i]; f[i + 8] = hi[i]; }
  return f;
}
// B: 32x16 f16 staged transposed in LDS as [n][k].
// lanes 0-15: N=lane, K=0..7,16..23 ; lanes 16-31: N=lane-16, K=8..15,24..31
template <int LDK>
__device__ __forceinline__ v16h frag_b(const half_t* base) {
  const int lane = threadIdx.x & 31;
  const half_t* p = base + (lane & 15) * LDK + ((lane >> 4) << 3);
  v8h lo = *(const v8h*)(p);
  v8h hi = *(const v8h*)(p + 16);
  v16h f;
#pragma unroll
  for (int i = 0; i < 8; ++i) { f[i] = lo[i]; f[i + 8] = hi[i]; }
  return f;
}

// ---------------- generic WMMA GEMM: C[M,N] = A[M,K] * Bt[N,K]^T -------------
// K-step 64 -> two v_wmma_f32_16x16x32_f16 per barrier pair. Ktot % 64 == 0.
// MODE 0: f32 out = acc + bias[n]                     (GRU gates)
// MODE 1: f16 out = relu(acc + bias[n])               (1x1 conv a)
// MODE 2: f16 out = relu(acc + bias[n] + res[m][n])   (1x1 conv b + residual)
// MODE 3: f16 out = acc + bias[n]                     (prediction heads)
template <int MODE>
__global__ __launch_bounds__(256) void gemm_f16_kernel(
    const half_t* __restrict__ A, long a_base, int a_stride,
    const half_t* __restrict__ Bt, const float* __restrict__ bias,
    const half_t* __restrict__ res, void* __restrict__ out,
    int N, int Ktot) {
  __shared__ __align__(16) half_t sA[MT * LDA];   // 64 x 64 (+8 pad)
  __shared__ __align__(16) half_t sB[NT * LDA];   // 32 x 64 (+8 pad)
  const int tid  = threadIdx.x;
  const int lane = tid & 31, wave = tid >> 5;
  const int wm = wave & 3, wn = wave >> 2;
  const long bm = (long)blockIdx.x * MT;
  const int  bn = blockIdx.y * NT;
  // A staging: 64 rows x 64 cols; thread -> (row = tid/4, col = (tid%4)*16)
  const int arow = tid >> 2;
  const int acol = (tid & 3) << 4;
  // B staging: 32 rows x 64 cols; thread -> (row = tid/8, col = (tid%8)*8)
  const int brow = tid >> 3;
  const int bcol = (tid & 7) << 3;
  v8f acc = {};
  for (int kt = 0; kt < Ktot; kt += KT2) {
    __syncthreads();
    const half_t* asrc = A + a_base + (bm + arow) * (long)a_stride + kt + acol;
    *(v8h*)(sA + arow * LDA + acol)     = *(const v8h*)(asrc);
    *(v8h*)(sA + arow * LDA + acol + 8) = *(const v8h*)(asrc + 8);
    if (kt + KT2 < Ktot) __builtin_prefetch(asrc + KT2, 0, 0);
    *(v8h*)(sB + brow * LDA + bcol) =
        *(const v8h*)(Bt + (long)(bn + brow) * Ktot + kt + bcol);
    __syncthreads();
    v16h fa0 = frag_a<LDA>(sA + wm * 16 * LDA);
    v16h fb0 = frag_b<LDA>(sB + wn * 16 * LDA);
    acc = __builtin_amdgcn_wmma_f32_16x16x32_f16(false, fa0, false, fb0,
                                                 (short)0, acc, false, false);
    v16h fa1 = frag_a<LDA>(sA + wm * 16 * LDA + 32);
    v16h fb1 = frag_b<LDA>(sB + wn * 16 * LDA + 32);
    acc = __builtin_amdgcn_wmma_f32_16x16x32_f16(false, fa1, false, fb1,
                                                 (short)0, acc, false, false);
  }
  const int  n  = bn + wn * 16 + (lane & 15);
  const long m0 = bm + wm * 16 + ((lane >> 4) << 3);
  const float bv = bias ? bias[n] : 0.0f;
#pragma unroll
  for (int v = 0; v < 8; ++v) {
    const long m = m0 + v;
    float val = acc[v] + bv;
    if (MODE == 0) {
      ((float*)out)[m * N + n] = val;
    } else if (MODE == 1) {
      ((half_t*)out)[m * N + n] = (half_t)fmaxf(val, 0.0f);
    } else if (MODE == 2) {
      val += (float)res[m * N + n];
      ((half_t*)out)[m * N + n] = (half_t)fmaxf(val, 0.0f);
    } else {
      ((half_t*)out)[m * N + n] = (half_t)val;
    }
  }
}

// ------------- encoder conv 5x5 stride2 pad2, 3->512, implicit GEMM ----------
// K = 75 zero-padded to 96 (3 x 32). Output f16 [16384][512], relu+bias.
__global__ __launch_bounds__(256) void enc_conv_kernel(
    const float* __restrict__ x, const half_t* __restrict__ Wt,
    const float* __restrict__ bias, half_t* __restrict__ out, int img_base) {
  __shared__ __align__(16) half_t sA[MT * LDE];   // 64 x 32 (+8 pad)
  __shared__ __align__(16) half_t sB[NT * LDE];   // 32 x 32 (+8 pad)
  const int tid  = threadIdx.x;
  const int lane = tid & 31, wave = tid >> 5;
  const int wm = wave & 3, wn = wave >> 2;
  const int bm = blockIdx.x * MT;
  const int bn = blockIdx.y * NT;
  const int aidx = tid * 8;
  const int ar = aidx >> 5, ac = aidx & 31;        // row 0..63, col 0/8/16/24
  const int p   = bm + ar;
  const int img = img_base + (p >> 8);
  const int op  = p & 255;
  const int oh = op >> 4, ow = op & 15;
  const int bidx = (tid & 127) * 8;
  const int brn = bidx >> 5, bc = bidx & 31;
  v8f acc = {};
  for (int kt = 0; kt < 96; kt += KTE) {
    __syncthreads();
#pragma unroll
    for (int j = 0; j < 8; ++j) {        // im2col gather with pad=2 bounds
      const int k = kt + ac + j;
      half_t v = (half_t)0.f;
      if (k < 75) {
        const int cch = k / 25;
        const int rem = k - cch * 25;
        const int kh = rem / 5;
        const int kw = rem - kh * 5;
        const int ih = oh * 2 - 2 + kh;
        const int iw = ow * 2 - 2 + kw;
        if ((unsigned)ih < 32u && (unsigned)iw < 32u)
          v = (half_t)x[(((long)img * 3 + cch) * 32 + ih) * 32 + iw];
      }
      sA[ar * LDE + ac + j] = v;
    }
    if (tid < 128)
      *(v8h*)(sB + brn * LDE + bc) =
          *(const v8h*)(Wt + (long)(bn + brn) * 96 + kt + bc);
    __syncthreads();
    v16h fa = frag_a<LDE>(sA + wm * 16 * LDE);
    v16h fb = frag_b<LDE>(sB + wn * 16 * LDE);
    acc = __builtin_amdgcn_wmma_f32_16x16x32_f16(false, fa, false, fb,
                                                 (short)0, acc, false, false);
  }
  const int  n  = bn + wn * 16 + (lane & 15);
  const long m0 = (long)bm + wm * 16 + ((lane >> 4) << 3);
  const float bv = bias[n];
#pragma unroll
  for (int v = 0; v < 8; ++v)
    out[(m0 + v) * 512 + n] = (half_t)fmaxf(acc[v] + bv, 0.0f);
}

// ------------- 3x3 conv pad1, 256->256, implicit GEMM, K = 9*256 ------------
// K-order: (tap = kh*3+kw, channel). Weights prepped to Wt[o][tap*256 + i].
__global__ __launch_bounds__(256) void conv3x3_kernel(
    const half_t* __restrict__ in, const half_t* __restrict__ Wt,
    const float* __restrict__ bias, half_t* __restrict__ out) {
  __shared__ __align__(16) half_t sA[MT * LDA];
  __shared__ __align__(16) half_t sB[NT * LDA];
  const int tid  = threadIdx.x;
  const int lane = tid & 31, wave = tid >> 5;
  const int wm = wave & 3, wn = wave >> 2;
  const int bm = blockIdx.x * MT;
  const int bn = blockIdx.y * NT;
  const int arow = tid >> 2;
  const int acol = (tid & 3) << 4;       // 16-half group stays in one tap
  const int brow = tid >> 3;
  const int bcol = (tid & 7) << 3;
  const int p    = bm + arow;
  const int imgL = p >> 8;
  const int op   = p & 255;
  const int oh = op >> 4, ow = op & 15;
  const int Ktot = 9 * 256;
  v8f acc = {};
  for (int kt = 0; kt < Ktot; kt += KT2) {
    __syncthreads();
    {
      const int k   = kt + acol;
      const int tap = k >> 8;
      const int i0  = k & 255;           // i0 multiple of 16 -> i0+16 <= 256
      const int kh = tap / 3;
      const int kw = tap - kh * 3;
      const int ih = oh - 1 + kh;
      const int iw = ow - 1 + kw;
      v8h v0 = {}, v1 = {};
      if ((unsigned)ih < 16u && (unsigned)iw < 16u) {
        const half_t* src =
            in + ((long)(imgL * 256 + ih * 16 + iw)) * 256 + i0;
        v0 = *(const v8h*)(src);
        v1 = *(const v8h*)(src + 8);
      }
      *(v8h*)(sA + arow * LDA + acol)     = v0;
      *(v8h*)(sA + arow * LDA + acol + 8) = v1;
    }
    *(v8h*)(sB + brow * LDA + bcol) =
        *(const v8h*)(Wt + (long)(bn + brow) * Ktot + kt + bcol);
    __syncthreads();
    v16h fa0 = frag_a<LDA>(sA + wm * 16 * LDA);
    v16h fb0 = frag_b<LDA>(sB + wn * 16 * LDA);
    acc = __builtin_amdgcn_wmma_f32_16x16x32_f16(false, fa0, false, fb0,
                                                 (short)0, acc, false, false);
    v16h fa1 = frag_a<LDA>(sA + wm * 16 * LDA + 32);
    v16h fb1 = frag_b<LDA>(sB + wn * 16 * LDA + 32);
    acc = __builtin_amdgcn_wmma_f32_16x16x32_f16(false, fa1, false, fb1,
                                                 (short)0, acc, false, false);
  }
  const int  n  = bn + wn * 16 + (lane & 15);
  const long m0 = (long)bm + wm * 16 + ((lane >> 4) << 3);
  const float bv = bias[n];
#pragma unroll
  for (int v = 0; v < 8; ++v)
    out[(m0 + v) * 256 + n] = (half_t)fmaxf(acc[v] + bv, 0.0f);
}

// ----------------------------- small helper kernels --------------------------
__global__ void cvt_f16_kernel(const float* __restrict__ s,
                               half_t* __restrict__ d, int n) {
  int i = blockIdx.x * 256 + threadIdx.x;
  if (i < n) d[i] = (half_t)s[i];
}

__global__ void prep_enc_w_kernel(const float* __restrict__ w,
                                  half_t* __restrict__ wt) {  // 512 x 96
  int i = blockIdx.x * 256 + threadIdx.x;
  if (i >= 512 * 96) return;
  int n = i / 96, k = i - n * 96;
  wt[i] = (k < 75) ? (half_t)w[n * 75 + k] : (half_t)0.f;
}

__global__ void prep_w2_kernel(const float* __restrict__ w,
                               half_t* __restrict__ wt) {  // (2*256) x 2304
  int i = blockIdx.x * 256 + threadIdx.x;
  if (i >= 512 * 2304) return;
  int ro = i / 2304, k = i - ro * 2304;
  int tap = k >> 8, ic = k & 255;
  wt[i] = (half_t)w[((long)(ro * 256 + ic)) * 9 + tap];  // [r][o][i][kh][kw]
}

__global__ void pool_kernel(const half_t* __restrict__ h, float* __restrict__ z,
                            half_t* __restrict__ zf16, int img_base) {
  int idx = blockIdx.x * 256 + threadIdx.x;  // 64 imgs * 512 ch
  if (idx >= 64 * 512) return;
  int imgL = idx >> 9, ch = idx & 511;
  float s = 0.f;
  for (int p = 0; p < 256; ++p)
    s += (float)h[((long)(imgL * 256 + p)) * 512 + ch];
  s *= (1.0f / 256.0f);
  long g = (long)(img_base + imgL) * 512 + ch;
  z[g] = s;
  zf16[g] = (half_t)s;
}

__global__ void gru_init_kernel(float* h, half_t* hf16) {
  int i = blockIdx.x * 256 + threadIdx.x;
  if (i < 64 * 256) { h[i] = 0.f; hf16[i] = (half_t)0.f; }
}

__global__ void gru_update_kernel(const float* __restrict__ gi,
                                  const float* __restrict__ gh,
                                  float* __restrict__ h,
                                  half_t* __restrict__ hf16) {
  int idx = blockIdx.x * 256 + threadIdx.x;  // 64*256
  if (idx >= 64 * 256) return;
  int b = idx >> 8, c = idx & 255;
  const float* gib = gi + b * 768;
  const float* ghb = gh + b * 768;
  float ir = gib[c], iz = gib[256 + c], in = gib[512 + c];
  float hr = ghb[c], hz = ghb[256 + c], hn = ghb[512 + c];
  float r = 1.f / (1.f + expf(-(ir + hr)));
  float z = 1.f / (1.f + expf(-(iz + hz)));
  float n = tanhf(in + r * hn);
  float hv = (1.f - z) * n + z * h[idx];
  h[idx] = hv;
  hf16[idx] = (half_t)hv;
}

__global__ void scores_kernel(const half_t* __restrict__ zf16,
                              const half_t* __restrict__ preds,
                              float* __restrict__ scores) {
  int idx = blockIdx.x * 256 + threadIdx.x;  // (i*64 + b)*5 + k
  if (idx >= 64 * 64 * 5) return;
  int k = idx % 5;
  int b = (idx / 5) & 63;
  int i = idx / (5 * 64);
  const half_t* za = zf16 + (long)(i * 9 + 4 + k) * 512;
  const half_t* pa = preds + (long)b * 2560 + k * 512;
  float s = 0.f;
  for (int d = 0; d < 512; d += 8) {
    v8h a = *(const v8h*)(za + d);
    v8h q = *(const v8h*)(pa + d);
#pragma unroll
    for (int j = 0; j < 8; ++j) s += (float)a[j] * (float)q[j];
  }
  scores[idx] = s;
}

__global__ __launch_bounds__(320) void loss_kernel(
    const float* __restrict__ scores, float* __restrict__ out) {
  __shared__ float sl[320];
  __shared__ float sa[320];
  int tid = threadIdx.x;           // one thread per (i, k); 64*5 = 320
  int i = tid / 5, k = tid % 5;
  float m = -1e30f;
  int am = 0;
  for (int b = 0; b < 64; ++b) {
    float yv = expf(scores[(i * 64 + b) * 5 + k]);
    if (yv > m) { m = yv; am = b; }
  }
  float s = 0.f;
  for (int b = 0; b < 64; ++b)
    s += expf(expf(scores[(i * 64 + b) * 5 + k]) - m);
  float lse = m + logf(s);
  float yi = expf(scores[(i * 64 + i) * 5 + k]);
  sl[tid] = -(yi - lse);
  sa[tid] = (am == i) ? 1.f : 0.f;
  __syncthreads();
  if (tid == 0) {
    float L = 0.f, A = 0.f;
    for (int t = 0; t < 320; ++t) { L += sl[t]; A += sa[t]; }
    out[0] = L / 320.f;
    out[1] = A / 320.f;
  }
}

// ------------------------------- host side -----------------------------------
extern "C" void kernel_launch(void* const* d_in, const int* in_sizes, int n_in,
                              void* d_out, int out_size, void* d_ws,
                              size_t ws_size, hipStream_t stream) {
  const float* x        = (const float*)d_in[0];
  const float* enc_w    = (const float*)d_in[1];
  const float* enc_b    = (const float*)d_in[2];
  const float* res_w1   = (const float*)d_in[3];
  const float* res_b1   = (const float*)d_in[4];
  const float* res_w2   = (const float*)d_in[5];
  const float* res_b2   = (const float*)d_in[6];
  const float* res_w3   = (const float*)d_in[7];
  const float* res_b3   = (const float*)d_in[8];
  const float* gru_w_ih = (const float*)d_in[9];
  const float* gru_w_hh = (const float*)d_in[10];
  const float* gru_b_ih = (const float*)d_in[11];
  const float* gru_b_hh = (const float*)d_in[12];
  const float* wk_w     = (const float*)d_in[13];
  const float* wk_b     = (const float*)d_in[14];

  char* ws = (char*)d_ws;
  size_t off = 0;
  auto alloc = [&](size_t bytes) -> void* {
    void* p = ws + off;
    off = (off + bytes + 255) & ~(size_t)255;
    return p;
  };
  half_t* enc_wt = (half_t*)alloc(512 * 96 * sizeof(half_t));
  half_t* w1f    = (half_t*)alloc(2 * 256 * 512 * sizeof(half_t));
  half_t* w2t    = (half_t*)alloc(2 * 256 * 2304 * sizeof(half_t));
  half_t* w3f    = (half_t*)alloc(2 * 512 * 256 * sizeof(half_t));
  half_t* wihf   = (half_t*)alloc(768 * 512 * sizeof(half_t));
  half_t* whhf   = (half_t*)alloc(768 * 256 * sizeof(half_t));
  half_t* wkf    = (half_t*)alloc(2560 * 256 * sizeof(half_t));
  half_t* bufA   = (half_t*)alloc((size_t)16384 * 512 * sizeof(half_t));
  half_t* bufB   = (half_t*)alloc((size_t)16384 * 512 * sizeof(half_t));
  half_t* y1     = (half_t*)alloc((size_t)16384 * 256 * sizeof(half_t));
  half_t* y2     = (half_t*)alloc((size_t)16384 * 256 * sizeof(half_t));
  float*  zf32   = (float*)alloc(576 * 512 * sizeof(float));
  half_t* zf16   = (half_t*)alloc(576 * 512 * sizeof(half_t));
  float*  hf32   = (float*)alloc(64 * 256 * sizeof(float));
  half_t* hf16   = (half_t*)alloc(64 * 256 * sizeof(half_t));
  float*  gi     = (float*)alloc(64 * 768 * sizeof(float));
  float*  gh     = (float*)alloc(64 * 768 * sizeof(float));
  half_t* preds  = (half_t*)alloc(64 * 2560 * sizeof(half_t));
  float*  scores = (float*)alloc(64 * 64 * 5 * sizeof(float));
  (void)ws_size; (void)in_sizes; (void)n_in; (void)out_size; (void)zf32;

  // ---- weight prep (f32 -> f16, with layout transforms where needed) ----
  cvt_f16_kernel<<<(2 * 256 * 512 + 255) / 256, 256, 0, stream>>>(res_w1, w1f, 2 * 256 * 512);
  prep_w2_kernel<<<(512 * 2304 + 255) / 256, 256, 0, stream>>>(res_w2, w2t);
  cvt_f16_kernel<<<(2 * 512 * 256 + 255) / 256, 256, 0, stream>>>(res_w3, w3f, 2 * 512 * 256);
  prep_enc_w_kernel<<<(512 * 96 + 255) / 256, 256, 0, stream>>>(enc_w, enc_wt);
  cvt_f16_kernel<<<(768 * 512 + 255) / 256, 256, 0, stream>>>(gru_w_ih, wihf, 768 * 512);
  cvt_f16_kernel<<<(768 * 256 + 255) / 256, 256, 0, stream>>>(gru_w_hh, whhf, 768 * 256);
  cvt_f16_kernel<<<(2560 * 256 + 255) / 256, 256, 0, stream>>>(wk_w, wkf, 2560 * 256);

  // ---- encoder + residual stack, 9 chunks of 64 images (L2-resident) ----
  for (int c = 0; c < 9; ++c) {
    const int img_base = c * 64;
    enc_conv_kernel<<<dim3(256, 16), 256, 0, stream>>>(x, enc_wt, enc_b, bufA, img_base);
    half_t* hcur = bufA;
    half_t* hnext = bufB;
    for (int r = 0; r < 2; ++r) {
      gemm_f16_kernel<1><<<dim3(256, 8), 256, 0, stream>>>(
          hcur, 0L, 512, w1f + (size_t)r * 256 * 512, res_b1 + r * 256,
          nullptr, y1, 256, 512);
      conv3x3_kernel<<<dim3(256, 8), 256, 0, stream>>>(
          y1, w2t + (size_t)r * 256 * 2304, res_b2 + r * 256, y2);
      gemm_f16_kernel<2><<<dim3(256, 16), 256, 0, stream>>>(
          y2, 0L, 256, w3f + (size_t)r * 512 * 256, res_b3 + r * 512,
          hcur, hnext, 512, 256);
      half_t* t = hcur; hcur = hnext; hnext = t;
    }
    pool_kernel<<<(64 * 512) / 256, 256, 0, stream>>>(hcur, zf32, zf16, img_base);
  }

  // ---- GRU over t = 0..3 ----
  gru_init_kernel<<<(64 * 256) / 256, 256, 0, stream>>>(hf32, hf16);
  for (int s = 0; s < 4; ++s) {
    gemm_f16_kernel<0><<<dim3(1, 24), 256, 0, stream>>>(
        zf16, (long)s * 512, 9 * 512, wihf, gru_b_ih, nullptr, gi, 768, 512);
    gemm_f16_kernel<0><<<dim3(1, 24), 256, 0, stream>>>(
        hf16, 0L, 256, whhf, gru_b_hh, nullptr, gh, 768, 256);
    gru_update_kernel<<<(64 * 256) / 256, 256, 0, stream>>>(gi, gh, hf32, hf16);
  }

  // ---- prediction heads: preds[b][k*512+d] = ct[b] . wk_w[k][d] + wk_b ----
  gemm_f16_kernel<3><<<dim3(1, 80), 256, 0, stream>>>(
      hf16, 0L, 256, wkf, wk_b, nullptr, preds, 2560, 256);

  // ---- scores + InfoNCE loss / accuracy ----
  scores_kernel<<<(64 * 64 * 5 + 255) / 256, 256, 0, stream>>>(zf16, preds, scores);
  loss_kernel<<<1, 320, 0, stream>>>(scores, (float*)d_out);
}
